// PaTHAttention_56057913148045
// MI455X (gfx1250) — matrix operand order repaired and checked by reading, not compiled, not run on hardware
//
#include <hip/hip_runtime.h>
#include <math.h>
#include <stdint.h>

// ---------------------------------------------------------------------------
// PaTH attention forward for MI455X (gfx1250, wave32, WMMA bf16,
// async global->LDS tile staging). B=1, T=1024, D=2048, H=32, hd=64.
// ---------------------------------------------------------------------------

typedef __bf16 bf16;
typedef __attribute__((ext_vector_type(16))) bf16  v16bf;
typedef __attribute__((ext_vector_type(8)))  float v8f;

#define T_LEN 1024
#define D_MOD 2048
#define NHEAD 32
#define HDIM  64
#define CHUNK 64
#define NCHUNK (T_LEN / CHUNK)
#define SCALE 0.125f /* hd^-0.5 */

__device__ __forceinline__ v8f wmma_bf16(v16bf a, v16bf b, v8f c) {
  // D = A(16x32 bf16) * B(32x16 bf16) + C(16x16 f32)
  return __builtin_amdgcn_wmma_f32_16x16x32_bf16(false, a, false, b,
                                                 (short)0, c, false, false);
}

// CDNA5 async global->LDS copy (ASYNCcnt-tracked). LDS operand is the 32-bit
// LDS byte offset (low 32 bits of a generic LDS pointer ARE the LDS offset).
__device__ __forceinline__ void async_g2l_b128(uint32_t lds_off,
                                               const void* gaddr) {
  unsigned long long ga = (unsigned long long)(uintptr_t)gaddr;
  asm volatile("global_load_async_to_lds_b128 %0, %1, off"
               :
               : "v"(lds_off), "v"(ga)
               : "memory");
}
__device__ __forceinline__ void wait_async0() {
  asm volatile("s_wait_asynccnt 0x0" ::: "memory");
}

// --- fragment gathers (ISA 7.12.2 layouts) ---------------------------------
// A fragment from row-major f32 X[m][k] (ld floats)
__device__ __forceinline__ v16bf afrag_f32(const float* X, int ld, int m0,
                                           int k0, int lane) {
  int m = lane & 15, hi = lane >> 4;
  const float* row = X + (size_t)(m0 + m) * ld;
  v16bf a;
#pragma unroll
  for (int j = 0; j < 8; ++j) {
    int kk = k0 + ((j < 4) ? (hi * 8 + 2 * j) : (16 + hi * 8 + 2 * (j - 4)));
    a[2 * j]     = (bf16)row[kk];
    a[2 * j + 1] = (bf16)row[kk + 1];
  }
  return a;
}
// A fragment from row-major bf16 X[m][k]
__device__ __forceinline__ v16bf afrag_bf16(const bf16* X, int ld, int m0,
                                            int k0, int lane) {
  int m = lane & 15, hi = lane >> 4;
  const bf16* row = X + (size_t)(m0 + m) * ld;
  v16bf a;
#pragma unroll
  for (int j = 0; j < 8; ++j) {
    int kk = k0 + ((j < 4) ? (hi * 8 + 2 * j) : (16 + hi * 8 + 2 * (j - 4)));
    a[2 * j]     = row[kk];
    a[2 * j + 1] = row[kk + 1];
  }
  return a;
}
// B fragment from row-major f32 X[k][n] (ld floats): same K order as A
// (lanes 0-15 K=0..7,16..23; lanes 16-31 K=8..15,24..31)
__device__ __forceinline__ v16bf bfrag_f32(const float* X, int ld, int k0,
                                           int n0, int lane) {
  int n = lane & 15, hi = lane >> 4;
  v16bf b;
#pragma unroll
  for (int j = 0; j < 8; ++j) {
    int kk = k0 + ((j < 4) ? (hi * 8 + 2 * j) : (16 + hi * 8 + 2 * (j - 4)));
    b[2 * j]     = (bf16)X[(size_t)kk * ld + n0 + n];
    b[2 * j + 1] = (bf16)X[(size_t)(kk + 1) * ld + n0 + n];
  }
  return b;
}
// B fragment from *transposed* storage Y[n][k] (B[k][n] = Y[n][k])
__device__ __forceinline__ v16bf bfragT_f32(const float* Y, int ld, int k0,
                                            int n0, int lane) {
  int n = lane & 15, hi = lane >> 4;
  const float* row = Y + (size_t)(n0 + n) * ld;
  v16bf b;
#pragma unroll
  for (int j = 0; j < 8; ++j) {
    int kk = k0 + ((j < 4) ? (hi * 8 + 2 * j) : (16 + hi * 8 + 2 * (j - 4)));
    b[2 * j]     = (bf16)row[kk];
    b[2 * j + 1] = (bf16)row[kk + 1];
  }
  return b;
}

// ---------------------------------------------------------------------------
// Generic tiled GEMM: C[M,N] = act(A[M,K] @ B[K,N] (+bias)), f32 in/out,
// bf16 WMMA inner product, async LDS staging. Tile 64x128, 8 wave32 waves,
// each wave owns a 2x2 grid of 16x16 accumulators.
// mode: 0=none, 1=2*sigmoid(+bias), 2=logsigmoid(+bias).
// ---------------------------------------------------------------------------
__global__ __launch_bounds__(256) void gemm_kernel(
    const float* __restrict__ A, const float* __restrict__ B,
    const float* __restrict__ bias, float* __restrict__ C, int M, int N,
    int K, int mode) {
  __shared__ float As[64][32];
  __shared__ float Bs[32][128];
  const int tid = threadIdx.x;
  const int lane = tid & 31;
  const int wave = tid >> 5;
  const int m_off = (wave >> 2) * 32;  // 0 / 32
  const int n_off = (wave & 3) * 32;   // 0..96
  const int bm = blockIdx.y * 64;
  const int bn = blockIdx.x * 128;

  v8f acc[2][2];
#pragma unroll
  for (int a = 0; a < 2; ++a)
#pragma unroll
    for (int b = 0; b < 2; ++b)
#pragma unroll
      for (int e = 0; e < 8; ++e) acc[a][b][e] = 0.f;

  for (int k0 = 0; k0 < K; k0 += 32) {
    // A tile: 64x32 f32 = 512 16B chunks, async straight into LDS
    for (int idx = tid; idx < 512; idx += 256) {
      int r = idx >> 3, c4 = (idx & 7) * 4;
      async_g2l_b128((uint32_t)(uintptr_t)&As[r][c4],
                     &A[(size_t)(bm + r) * K + k0 + c4]);
    }
    // B tile: 32x128 f32 = 1024 16B chunks
    for (int idx = tid; idx < 1024; idx += 256) {
      int r = idx >> 5, c4 = (idx & 31) * 4;
      if (bn + c4 + 3 < N) {
        async_g2l_b128((uint32_t)(uintptr_t)&Bs[r][c4],
                       &B[(size_t)(k0 + r) * N + bn + c4]);
      } else {
#pragma unroll
        for (int u = 0; u < 4; ++u)
          Bs[r][c4 + u] =
              (bn + c4 + u < N) ? B[(size_t)(k0 + r) * N + bn + c4 + u] : 0.f;
      }
    }
    wait_async0();
    __syncthreads();
    v16bf af[2], bfv[2];
    af[0] = afrag_f32(&As[0][0], 32, m_off, 0, lane);
    af[1] = afrag_f32(&As[0][0], 32, m_off + 16, 0, lane);
    bfv[0] = bfrag_f32(&Bs[0][0], 128, 0, n_off, lane);
    bfv[1] = bfrag_f32(&Bs[0][0], 128, 0, n_off + 16, lane);
#pragma unroll
    for (int mt = 0; mt < 2; ++mt)
#pragma unroll
      for (int nt = 0; nt < 2; ++nt)
        acc[mt][nt] = wmma_bf16(af[mt], bfv[nt], acc[mt][nt]);
    __syncthreads();
  }
  const int lm = lane & 15, hi = lane >> 4;
#pragma unroll
  for (int mt = 0; mt < 2; ++mt)
#pragma unroll
    for (int nt = 0; nt < 2; ++nt)
#pragma unroll
      for (int e = 0; e < 8; ++e) {
        int gm = bm + m_off + mt * 16 + e + 8 * hi;
        int gn = bn + n_off + nt * 16 + lm;
        if (gm < M && gn < N) {
          float x = acc[mt][nt][e];
          if (mode == 1) {
            x = 2.f / (1.f + __expf(-(x + bias[gn])));
          } else if (mode == 2) {
            float z = x + bias[gn];
            x = fminf(z, 0.f) - log1pf(__expf(-fabsf(z)));
          }
          C[(size_t)gm * N + gn] = x;
        }
      }
}

// ---------------------------------------------------------------------------
// Depthwise causal conv(K=3) + SiLU + per-(t,head) L2 norm; output head-major
// wh[H][T][64] (f32). One wave per (t,h); lane covers 2 channels.
// ---------------------------------------------------------------------------
__global__ __launch_bounds__(32) void conv_silu_norm_kernel(
    const float* __restrict__ wfull, const float* __restrict__ conv_w,
    float* __restrict__ wh) {
  int bid = blockIdx.x;            // t*H + h
  int t = bid >> 5, h = bid & 31;  // H = 32
  int lane = threadIdx.x;
  float vals[2];
#pragma unroll
  for (int u = 0; u < 2; ++u) {
    int c = h * HDIM + lane + u * 32;
    float acc = 0.f;
#pragma unroll
    for (int j = 0; j < 3; ++j) {
      int ts = t + j - 2;
      if (ts >= 0) acc += wfull[(size_t)ts * D_MOD + c] * conv_w[c * 3 + j];
    }
    vals[u] = acc / (1.f + __expf(-acc));  // silu
  }
  float ss = vals[0] * vals[0] + vals[1] * vals[1];
#pragma unroll
  for (int m = 16; m >= 1; m >>= 1) ss += __shfl_xor(ss, m, 32);
  float inv = rsqrtf(ss);
  size_t base = ((size_t)h * T_LEN + t) * HDIM;
  wh[base + lane] = vals[0] * inv;
  wh[base + lane + 32] = vals[1] * inv;
}

// q/k/v [T][D] -> head-major [H][T][64]
__global__ __launch_bounds__(256) void pack_heads_kernel(
    const float* __restrict__ q, const float* __restrict__ k,
    const float* __restrict__ v, float* __restrict__ qh,
    float* __restrict__ kh, float* __restrict__ vh) {
  size_t idx = (size_t)blockIdx.x * 256 + threadIdx.x;
  if (idx >= (size_t)T_LEN * D_MOD) return;
  int t = idx / D_MOD, c = idx % D_MOD;
  int h = c / HDIM, d = c % HDIM;
  size_t o = ((size_t)h * T_LEN + t) * HDIM + d;
  qh[o] = q[idx];
  kh[o] = k[idx];
  vh[o] = v[idx];
}

// beta[T][H] -> betaT[H][T]; g[T][H] (already log-sigmoid) -> G[H][T] cumsum
__global__ void gates_kernel(const float* __restrict__ beta,
                             const float* __restrict__ g,
                             float* __restrict__ betaT, float* __restrict__ G) {
  int h = threadIdx.x;
  if (h >= NHEAD) return;
  float acc = 0.f;
  for (int t = 0; t < T_LEN; ++t) {
    acc += g[t * NHEAD + h];
    G[h * T_LEN + t] = acc;
    betaT[h * T_LEN + t] = beta[t * NHEAD + h];
  }
}

// ---------------------------------------------------------------------------
// Chunked PaTH attention. One block (8 waves) per head. Per 64-row chunk:
//   1) descending rank-1 sweep builds qtilde (back-transformed queries)
//   2) flash loop over prior 64-col blocks: WMMA scores qtilde * Khat^T,
//      gated softmax, WMMA P*V (V async-staged in LDS, overlapped with WMMA)
//   3) ascending rank-1 sweep builds intra-chunk keys, intra scores and the
//      chunk product Pc; intra flash step
//   4) WMMA update Khat <- Khat * Pc^T, append chunk keys, write O
// Khat (per-head [T][64] f32) lives in global scratch; 64x64 panels in LDS.
// ---------------------------------------------------------------------------
__global__ __launch_bounds__(256) void path_attn_kernel(
    const float* __restrict__ qh, const float* __restrict__ kh,
    const float* __restrict__ vh, const float* __restrict__ wh,
    const float* __restrict__ betaT, const float* __restrict__ G,
    float* __restrict__ khat_ws, float* __restrict__ o) {
  const int h = blockIdx.x;
  const int tid = threadIdx.x;
  const int lane = tid & 31;
  const int wave = tid >> 5;
  const int lm = lane & 15, hi = lane >> 4;
  const int mi = wave & 3;             // this wave's m-tile (rows)
  const int nbase = (wave >> 2) * 2;   // this wave's two n-tiles

  __shared__ float Qc[CHUNK][HDIM];    // raw queries of chunk
  __shared__ float Qt[CHUNK][HDIM];    // back-transformed queries
  __shared__ float Wc[CHUNK][HDIM];    // householder directions
  __shared__ float Kc[CHUNK][HDIM];    // evolving intra-chunk keys
  __shared__ float Mp[CHUNK][HDIM];    // chunk product Pc
  __shared__ float Vs[CHUNK][HDIM];    // async-staged V block
  __shared__ float sblk[CHUNK][CHUNK]; // score block
  __shared__ bf16  Pblk[CHUNK][CHUNK]; // softmax probs (bf16 for WMMA)
  __shared__ float betac[CHUNK], dred[CHUNK], dcol[CHUNK];
  __shared__ float mrow[CHUNK], lrow[CHUNK], arow[CHUNK];
  __shared__ float gcol[CHUNK], grow[CHUNK];

  const float* qH = qh + (size_t)h * T_LEN * HDIM;
  const float* kH = kh + (size_t)h * T_LEN * HDIM;
  const float* vH = vh + (size_t)h * T_LEN * HDIM;
  const float* wH = wh + (size_t)h * T_LEN * HDIM;
  const float* gH = G + (size_t)h * T_LEN;
  const float* bH = betaT + (size_t)h * T_LEN;
  float* KH = khat_ws + (size_t)h * T_LEN * HDIM;

  v8f oacc[2];

  for (int rc = 0; rc < NCHUNK; ++rc) {
    const int s0 = rc * CHUNK;
    // ---- stage chunk data ----
    for (int idx = tid; idx < CHUNK * HDIM; idx += 256) {
      int r = idx >> 6, c = idx & 63;
      float qv = qH[(size_t)(s0 + r) * HDIM + c];
      Qc[r][c] = qv;
      Qt[r][c] = qv;
      Wc[r][c] = wH[(size_t)(s0 + r) * HDIM + c];
      Mp[r][c] = (r == c) ? 1.f : 0.f;
    }
    if (tid < CHUNK) {
      betac[tid] = bH[s0 + tid];
      grow[tid] = gH[s0 + tid];
      mrow[tid] = -3.0e38f;
      lrow[tid] = 0.f;
    }
#pragma unroll
    for (int t2 = 0; t2 < 2; ++t2)
#pragma unroll
      for (int e = 0; e < 8; ++e) oacc[t2][e] = 0.f;
    __syncthreads();

    // ---- descending sweep: Qt[j] <- H_i Qt[j] for i = 63..0, rows j>=i ----
    for (int i = CHUNK - 1; i >= 0; --i) {
      if (tid < CHUNK && tid >= i) {
        float d = 0.f;
        for (int c = 0; c < HDIM; ++c) d += Qt[tid][c] * Wc[i][c];
        dred[tid] = d;
      }
      __syncthreads();
      for (int idx = tid; idx < CHUNK * HDIM; idx += 256) {
        int r = idx >> 6, c = idx & 63;
        if (r >= i) Qt[r][c] -= betac[i] * dred[r] * Wc[i][c];
      }
      __syncthreads();
    }

    // ---- flash over prior key blocks ----
    for (int sb = 0; sb < rc; ++sb) {
      // async-stage V block (overlaps with score WMMAs below)
      for (int idx = tid; idx < 1024; idx += 256) {
        int r = idx >> 4, c4 = (idx & 15) * 4;
        async_g2l_b128((uint32_t)(uintptr_t)&Vs[r][c4],
                       &vH[(size_t)(sb * CHUNK + r) * HDIM + c4]);
      }
      // prefetch current Khat panel into caches (global_prefetch_b8)
      __builtin_prefetch(KH + (size_t)sb * CHUNK * HDIM + tid * 16, 0, 1);

      v8f acc[2];
#pragma unroll
      for (int t2 = 0; t2 < 2; ++t2)
#pragma unroll
        for (int e = 0; e < 8; ++e) acc[t2][e] = 0.f;
#pragma unroll
      for (int ks = 0; ks < 2; ++ks) {
        v16bf af = afrag_f32(&Qt[0][0], HDIM, mi * 16, ks * 32, lane);
#pragma unroll
        for (int t2 = 0; t2 < 2; ++t2) {
          v16bf bfv = bfragT_f32(KH + (size_t)sb * CHUNK * HDIM, HDIM,
                                 ks * 32, (nbase + t2) * 16, lane);
          acc[t2] = wmma_bf16(af, bfv, acc[t2]);
        }
      }
#pragma unroll
      for (int t2 = 0; t2 < 2; ++t2)
#pragma unroll
        for (int e = 0; e < 8; ++e)
          sblk[mi * 16 + e + 8 * hi][(nbase + t2) * 16 + lm] = acc[t2][e];
      if (tid < CHUNK) gcol[tid] = gH[sb * CHUNK + tid];
      wait_async0();
      __syncthreads();
      // gated flash-softmax row update
      if (tid < CHUNK) {
        int r = tid;
        float mo = mrow[r], bm = -3.0e38f;
        for (int c = 0; c < CHUNK; ++c) {
          float lg = sblk[r][c] * SCALE + grow[r] - gcol[c];
          sblk[r][c] = lg;
          bm = fmaxf(bm, lg);
        }
        float mn = fmaxf(mo, bm);
        float al = __expf(mo - mn);
        float s = 0.f;
        for (int c = 0; c < CHUNK; ++c) {
          float p = __expf(sblk[r][c] - mn);
          Pblk[r][c] = (bf16)p;
          s += p;
        }
        mrow[r] = mn;
        lrow[r] = lrow[r] * al + s;
        arow[r] = al;
      }
      __syncthreads();
      // rescale running O, accumulate P * V
#pragma unroll
      for (int t2 = 0; t2 < 2; ++t2)
#pragma unroll
        for (int e = 0; e < 8; ++e)
          oacc[t2][e] *= arow[mi * 16 + e + 8 * hi];
#pragma unroll
      for (int ks = 0; ks < 2; ++ks) {
        v16bf pf = afrag_bf16(&Pblk[0][0], CHUNK, mi * 16, ks * 32, lane);
#pragma unroll
        for (int t2 = 0; t2 < 2; ++t2) {
          v16bf vf = bfrag_f32(&Vs[0][0], HDIM, ks * 32,
                               (nbase + t2) * 16, lane);
          oacc[t2] = wmma_bf16(pf, vf, oacc[t2]);
        }
      }
      __syncthreads();
    }

    // ---- intra-chunk: ascending sweep (Kc, intra scores, Pc) ----
    for (int idx = tid; idx < CHUNK * CHUNK; idx += 256)
      sblk[idx >> 6][idx & 63] = -1e30f;  // strict-causal mask default
    // async-stage the diagonal V block while the sweep runs
    for (int idx = tid; idx < 1024; idx += 256) {
      int r = idx >> 4, c4 = (idx & 15) * 4;
      async_g2l_b128((uint32_t)(uintptr_t)&Vs[r][c4],
                     &vH[(size_t)(s0 + r) * HDIM + c4]);
    }
    __syncthreads();
    for (int i = 0; i < CHUNK; ++i) {
      if (tid < CHUNK) {
        float d = 0.f;  // column dot: (w_i^T Mp)[col]
        for (int r = 0; r < CHUNK; ++r) d += Wc[i][r] * Mp[r][tid];
        dcol[tid] = d;
        if (tid < i) {
          float d2 = 0.f;
          for (int c = 0; c < HDIM; ++c) d2 += Kc[tid][c] * Wc[i][c];
          dred[tid] = d2;
        }
      }
      __syncthreads();
      for (int idx = tid; idx < CHUNK * HDIM; idx += 256) {
        int r = idx >> 6, c = idx & 63;
        Mp[r][c] -= betac[i] * Wc[i][r] * dcol[c];
        if (r < i) Kc[r][c] -= betac[i] * dred[r] * Wc[i][c];
      }
      if (tid < HDIM) Kc[i][tid] = kH[(size_t)(s0 + i) * HDIM + tid];
      __syncthreads();
      if (tid <= i) {
        float d3 = 0.f;
        for (int c = 0; c < HDIM; ++c) d3 += Qc[i][c] * Kc[tid][c];
        sblk[i][tid] = d3;
      }
      __syncthreads();
    }

    // ---- intra flash step ----
    if (tid < CHUNK) gcol[tid] = grow[tid];
    wait_async0();
    __syncthreads();
    if (tid < CHUNK) {
      int r = tid;
      float mo = mrow[r], bm = -3.0e38f;
      for (int c = 0; c < CHUNK; ++c) {
        float lg = sblk[r][c] * SCALE + grow[r] - gcol[c];
        sblk[r][c] = lg;
        bm = fmaxf(bm, lg);
      }
      float mn = fmaxf(mo, bm);
      float al = __expf(mo - mn);
      float s = 0.f;
      for (int c = 0; c < CHUNK; ++c) {
        float p = __expf(sblk[r][c] - mn);
        Pblk[r][c] = (bf16)p;
        s += p;
      }
      mrow[r] = mn;
      lrow[r] = lrow[r] * al + s;
      arow[r] = al;
    }
    __syncthreads();
#pragma unroll
    for (int t2 = 0; t2 < 2; ++t2)
#pragma unroll
      for (int e = 0; e < 8; ++e) oacc[t2][e] *= arow[mi * 16 + e + 8 * hi];
#pragma unroll
    for (int ks = 0; ks < 2; ++ks) {
      v16bf pf = afrag_bf16(&Pblk[0][0], CHUNK, mi * 16, ks * 32, lane);
#pragma unroll
      for (int t2 = 0; t2 < 2; ++t2) {
        v16bf vf = bfrag_f32(&Vs[0][0], HDIM, ks * 32,
                             (nbase + t2) * 16, lane);
        oacc[t2] = wmma_bf16(pf, vf, oacc[t2]);
      }
    }
    __syncthreads();

    // ---- Khat <- Khat * Pc^T (per prior block), then append chunk keys ----
    for (int sb = 0; sb < rc; ++sb) {
      v8f up[2];
#pragma unroll
      for (int t2 = 0; t2 < 2; ++t2)
#pragma unroll
        for (int e = 0; e < 8; ++e) up[t2][e] = 0.f;
#pragma unroll
      for (int ks = 0; ks < 2; ++ks) {
        v16bf af = afrag_f32(KH, HDIM, sb * CHUNK + mi * 16, ks * 32, lane);
#pragma unroll
        for (int t2 = 0; t2 < 2; ++t2) {
          v16bf bfv = bfragT_f32(&Mp[0][0], HDIM, ks * 32,
                                 (nbase + t2) * 16, lane);
          up[t2] = wmma_bf16(af, bfv, up[t2]);
        }
      }
      __syncthreads();
#pragma unroll
      for (int t2 = 0; t2 < 2; ++t2)
#pragma unroll
        for (int e = 0; e < 8; ++e)
          KH[(size_t)(sb * CHUNK + mi * 16 + e + 8 * hi) * HDIM +
             (nbase + t2) * 16 + lm] = up[t2][e];
      __syncthreads();
    }
    for (int idx = tid; idx < CHUNK * HDIM; idx += 256) {
      int r = idx >> 6, c = idx & 63;
      KH[(size_t)(s0 + r) * HDIM + c] = Kc[r][c];
    }

    // ---- write O rows (normalize by flash denominator) ----
#pragma unroll
    for (int t2 = 0; t2 < 2; ++t2)
#pragma unroll
      for (int e = 0; e < 8; ++e) {
        int rr = mi * 16 + e + 8 * hi;
        int cc = (nbase + t2) * 16 + lm;
        o[(size_t)(s0 + rr) * D_MOD + h * HDIM + cc] = oacc[t2][e] / lrow[rr];
      }
    __syncthreads();
  }
}

// ---------------------------------------------------------------------------
extern "C" void kernel_launch(void* const* d_in, const int* in_sizes, int n_in,
                              void* d_out, int out_size, void* d_ws,
                              size_t ws_size, hipStream_t stream) {
  (void)in_sizes; (void)n_in; (void)out_size; (void)ws_size;
  const float* h    = (const float*)d_in[0];
  const float* Wq   = (const float*)d_in[1];
  const float* Wk   = (const float*)d_in[2];
  const float* Wv   = (const float*)d_in[3];
  const float* Ww1  = (const float*)d_in[4];
  const float* Ww2  = (const float*)d_in[5];
  const float* cw   = (const float*)d_in[6];
  const float* Wbt  = (const float*)d_in[7];
  const float* bbt  = (const float*)d_in[8];
  const float* Wg   = (const float*)d_in[9];
  const float* bg   = (const float*)d_in[10];
  const float* Wo   = (const float*)d_in[11];
  float* out = (float*)d_out;

  const size_t TD = (size_t)T_LEN * D_MOD;
  char* p = (char*)d_ws;
  float* q      = (float*)p; p += TD * 4;
  float* k      = (float*)p; p += TD * 4;
  float* v      = (float*)p; p += TD * 4;
  float* wfull  = (float*)p; p += TD * 4;
  float* obuf   = (float*)p; p += TD * 4;
  float* qhh    = (float*)p; p += TD * 4;
  float* khh    = (float*)p; p += TD * 4;
  float* vhh    = (float*)p; p += TD * 4;
  float* whh    = (float*)p; p += TD * 4;
  float* khat   = (float*)p; p += TD * 4;  // per-head transformed keys
  float* tmp32  = (float*)p; p += (size_t)T_LEN * 32 * 4;
  float* beta   = (float*)p; p += (size_t)T_LEN * NHEAD * 4;
  float* gbuf   = (float*)p; p += (size_t)T_LEN * NHEAD * 4;
  float* betaT  = (float*)p; p += (size_t)NHEAD * T_LEN * 4;
  float* Gc     = (float*)p; p += (size_t)NHEAD * T_LEN * 4;

  dim3 blk(256);
  dim3 gFull((D_MOD + 127) / 128, T_LEN / 64);   // 16 x 16
  dim3 gNarrow(1, T_LEN / 64);                   // N = 32

  gemm_kernel<<<gFull, blk, 0, stream>>>(h, Wq, nullptr, q, T_LEN, D_MOD, D_MOD, 0);
  gemm_kernel<<<gFull, blk, 0, stream>>>(h, Wk, nullptr, k, T_LEN, D_MOD, D_MOD, 0);
  gemm_kernel<<<gFull, blk, 0, stream>>>(h, Wv, nullptr, v, T_LEN, D_MOD, D_MOD, 0);
  gemm_kernel<<<gNarrow, blk, 0, stream>>>(h, Ww1, nullptr, tmp32, T_LEN, 32, D_MOD, 0);
  gemm_kernel<<<gFull, blk, 0, stream>>>(tmp32, Ww2, nullptr, wfull, T_LEN, D_MOD, 32, 0);
  gemm_kernel<<<gNarrow, blk, 0, stream>>>(h, Wbt, bbt, beta, T_LEN, NHEAD, D_MOD, 1);
  gemm_kernel<<<gNarrow, blk, 0, stream>>>(h, Wg, bg, gbuf, T_LEN, NHEAD, D_MOD, 2);

  conv_silu_norm_kernel<<<T_LEN * NHEAD, 32, 0, stream>>>(wfull, cw, whh);
  pack_heads_kernel<<<(int)(TD / 256), blk, 0, stream>>>(q, k, v, qhh, khh, vhh);
  gates_kernel<<<1, 32, 0, stream>>>(beta, gbuf, betaT, Gc);

  path_attn_kernel<<<NHEAD, blk, 0, stream>>>(qhh, khh, vhh, whh, betaT, Gc,
                                              khat, obuf);

  gemm_kernel<<<gFull, blk, 0, stream>>>(obuf, Wo, nullptr, out, T_LEN, D_MOD,
                                         D_MOD, 0);
}
